// LPModel_22617297780819
// MI455X (gfx1250) — matrix operand order repaired and checked by reading, not compiled, not run on hardware
//
#include <hip/hip_runtime.h>
#include <math.h>

// CDNA5 / gfx1250: wave32, WMMA 16x16x32 bf16 with f32 accumulate.
typedef __attribute__((ext_vector_type(16))) __bf16 v16bf;
typedef __attribute__((ext_vector_type(8)))  float  v8f;

static inline int div_up(int a, int b) { return (a + b - 1) / b; }

#define TPB 256

// ============================ device helpers ============================

__device__ __forceinline__ void atomicMaxF(float* addr, float v) {
  // standard two-sided int/uint trick (works for mixed signs, init to -3e38)
  if (v >= 0.0f) atomicMax((int*)addr, __float_as_int(v));
  else           atomicMin((unsigned int*)addr, __float_as_uint(v));
}

// ============================ weight packing ============================
// B-matrix 32x16 (KxN) bf16 layout: lane L (n = L&15) holds k = (L>>4)*16 + i
__global__ void k_pack_b(const float* __restrict__ W, __bf16* __restrict__ P, int K, int N) {
  int tid = blockIdx.x * blockDim.x + threadIdx.x;
  int ksteps = K >> 5;
  int total  = (N >> 4) * ksteps * 32;
  if (tid >= total) return;
  int lane  = tid & 31;
  int kstep = (tid >> 5) % ksteps;
  int ntile = (tid >> 5) / ksteps;
  int n     = ntile * 16 + (lane & 15);
  int kbase = kstep * 32 + (lane >> 4) * 16;
  v16bf out;
#pragma unroll
  for (int i = 0; i < 16; ++i)
    out[i] = (__bf16)W[(size_t)(kbase + i) * N + n];
  *(v16bf*)(P + (size_t)tid * 16) = out;
}

// ============================ WMMA GEMM ============================
// One wave computes a 16-row x 128-col block = 8 WMMA tiles sharing one A fragment.
// KSTEPS is a template constant so all 8 B-fragment addresses become one base pointer
// plus compile-time immediate offsets (t*KSTEPS*1024 bytes < 2^23) -> loads clause
// together into distinct registers and the 8 WMMAs issue back-to-back.
// A is f32 row-major, converted to bf16 inline (A-layout: lane L -> m=L&15, half=L>>4;
// slot i -> k=half*16+i).
// Y[M,N] = A(M,K) @ Bpack(K,N) + bias (+addend) (+relu).  K=KSTEPS*32, N%128==0.
template <int KSTEPS>
__global__ void k_gemm_wmma(const float* __restrict__ A, const __bf16* __restrict__ Bpack,
                            const float* __restrict__ bias, const float* __restrict__ addend,
                            float* __restrict__ Y, int M, int N, int relu) {
  const int K = KSTEPS * 32;
  int wid  = (blockIdx.x * blockDim.x + threadIdx.x) >> 5;
  int lane = threadIdx.x & 31;
  int ngroups = N >> 7;                       // groups of 8 n-tiles (128 cols)
  int mtiles  = M >> 4;
  if (wid >= mtiles * ngroups) return;        // wave-uniform: EXEC all-ones for WMMA
  int grp   = wid % ngroups;
  int mtile = wid / ngroups;
  int half  = lane >> 4;
  const float* arow = A + (size_t)(mtile * 16 + (lane & 15)) * K + half * 16;
  const v16bf* B = (const v16bf*)Bpack + (size_t)(grp * 8) * KSTEPS * 32 + lane;
  v8f acc[8] = {};
  for (int k = 0; k < KSTEPS; ++k) {
    const float4* p0 = (const float4*)(arow + (size_t)k * 32);
    const float4* p1 = (const float4*)(arow + (size_t)k * 32 + 8);
    float4 a0 = p0[0], a1 = p0[1];
    float4 a2 = p1[0], a3 = p1[1];
    v16bf bv[8];
#pragma unroll
    for (int t = 0; t < 8; ++t)
      bv[t] = B[(size_t)(t * KSTEPS + k) * 32];   // immediate offset: t*KSTEPS*1024 bytes
    v16bf av;
    av[0]  = (__bf16)a0.x; av[1]  = (__bf16)a0.y; av[2]  = (__bf16)a0.z; av[3]  = (__bf16)a0.w;
    av[4]  = (__bf16)a1.x; av[5]  = (__bf16)a1.y; av[6]  = (__bf16)a1.z; av[7]  = (__bf16)a1.w;
    av[8]  = (__bf16)a2.x; av[9]  = (__bf16)a2.y; av[10] = (__bf16)a2.z; av[11] = (__bf16)a2.w;
    av[12] = (__bf16)a3.x; av[13] = (__bf16)a3.y; av[14] = (__bf16)a3.z; av[15] = (__bf16)a3.w;
#pragma unroll
    for (int t = 0; t < 8; ++t)
      acc[t] = __builtin_amdgcn_wmma_f32_16x16x32_bf16(false, av, false, bv[t], (short)0,
                                                       acc[t], false, false);
  }
  // C/D layout: lane L, vgpr j -> m = half*8 + j, n = L&15
  int m0 = mtile * 16 + half * 8;
  int n0 = grp * 128 + (lane & 15);
#pragma unroll
  for (int t = 0; t < 8; ++t) {
    int n = n0 + t * 16;
    float bv = bias ? bias[n] : 0.0f;
#pragma unroll
    for (int j = 0; j < 8; ++j) {
      size_t o   = (size_t)(m0 + j) * N + n;
      float  val = acc[t][j] + bv;
      if (addend) val += addend[o];
      if (relu)   val = val > 0.0f ? val : 0.0f;
      Y[o] = val;
    }
  }
}

// ============================ elementwise / LN ============================

__global__ void k_fill(float* __restrict__ p, float v, int n) {
  int t = blockIdx.x * blockDim.x + threadIdx.x;
  if (t < n) p[t] = v;
}

// din==2 linear: Y[m,n] = b[n] + X[m,0]*W[0,n] + X[m,1]*W[1,n]
__global__ void k_lin2(const float* __restrict__ X, const float* __restrict__ W2,
                       const float* __restrict__ b, float* __restrict__ Y, int M, int N) {
  int tid = blockIdx.x * blockDim.x + threadIdx.x;
  if (tid >= M * N) return;
  int m = tid / N, n = tid - m * N;
  Y[tid] = b[n] + X[(size_t)m * 2] * W2[n] + X[(size_t)m * 2 + 1] * W2[N + n];
}

// LayerNorm over D=128, one wave per row, 4 elems per lane
__global__ void k_layernorm(const float* __restrict__ X, const float* __restrict__ g,
                            const float* __restrict__ b, float* __restrict__ Y, int M) {
  int wid  = (blockIdx.x * blockDim.x + threadIdx.x) >> 5;
  int lane = threadIdx.x & 31;
  if (wid >= M) return;
  float4 v = ((const float4*)(X + (size_t)wid * 128))[lane];
  float s = v.x + v.y + v.z + v.w;
#pragma unroll
  for (int off = 16; off > 0; off >>= 1) s += __shfl_xor(s, off, 32);
  float mu = s * (1.0f / 128.0f);
  float d0 = v.x - mu, d1 = v.y - mu, d2 = v.z - mu, d3 = v.w - mu;
  float q = d0 * d0 + d1 * d1 + d2 * d2 + d3 * d3;
#pragma unroll
  for (int off = 16; off > 0; off >>= 1) q += __shfl_xor(q, off, 32);
  float inv = rsqrtf(q * (1.0f / 128.0f) + 1e-5f);
  float4 gv = ((const float4*)g)[lane];
  float4 bv = ((const float4*)b)[lane];
  float4 o;
  o.x = d0 * inv * gv.x + bv.x;
  o.y = d1 * inv * gv.y + bv.y;
  o.z = d2 * inv * gv.z + bv.z;
  o.w = d3 * inv * gv.w + bv.w;
  ((float4*)(Y + (size_t)wid * 128))[lane] = o;
}

// out[e,:] = eX[e,:] + nX[u[e],:] + nX[v[e],:]
__global__ void k_combine3(const float* __restrict__ eX, const float* __restrict__ nX,
                           const int* __restrict__ u, const int* __restrict__ v,
                           float* __restrict__ out, int E) {
  int tid = blockIdx.x * blockDim.x + threadIdx.x;
  if (tid >= E * 128) return;
  int e = tid >> 7, d = tid & 127;
  out[tid] = eX[tid] + nX[(size_t)u[e] * 128 + d] + nX[(size_t)v[e] * 128 + d];
}

// ============================ attention (segment softmax) ============================
// node attention: Q = eQ[e]+nQ[u[e]], K = eK[e]+nK[v[e]], seg = u
__global__ void k_attn_score_node(const float* __restrict__ eQ, const float* __restrict__ eK,
                                  const float* __restrict__ nQ, const float* __restrict__ nK,
                                  const int* __restrict__ u, const int* __restrict__ v,
                                  float* __restrict__ score, float* __restrict__ segmax, int E) {
  int tid = blockIdx.x * blockDim.x + threadIdx.x;
  if (tid >= E * 8) return;
  int e = tid >> 3, hh = tid & 7;
  int s = u[e], d = v[e];
  const float* q1 = eQ + (size_t)e * 128 + hh * 16;
  const float* q2 = nQ + (size_t)s * 128 + hh * 16;
  const float* k1 = eK + (size_t)e * 128 + hh * 16;
  const float* k2 = nK + (size_t)d * 128 + hh * 16;
  float acc = 0.0f;
#pragma unroll
  for (int i = 0; i < 16; ++i) acc += (q1[i] + q2[i]) * (k1[i] + k2[i]);
  acc *= 0.25f;  // 1/sqrt(16)
  score[tid] = acc;
  atomicMaxF(&segmax[(size_t)s * 8 + hh], acc);
}

// pair attention: score = Qe[nsrc[p]] . Ke[ndst[p]], seg = nsrc
__global__ void k_attn_score_pair(const float* __restrict__ Qe, const float* __restrict__ Ke,
                                  const int* __restrict__ ps, const int* __restrict__ pd,
                                  float* __restrict__ score, float* __restrict__ segmax, int P) {
  int tid = blockIdx.x * blockDim.x + threadIdx.x;
  if (tid >= P * 8) return;
  int p = tid >> 3, hh = tid & 7;
  int a = ps[p], b = pd[p];
  const float* q = Qe + (size_t)a * 128 + hh * 16;
  const float* k = Ke + (size_t)b * 128 + hh * 16;
  float acc = 0.0f;
#pragma unroll
  for (int i = 0; i < 16; ++i) acc += q[i] * k[i];
  acc *= 0.25f;
  score[tid] = acc;
  atomicMaxF(&segmax[(size_t)a * 8 + hh], acc);
}

// in-place: score <- exp(score - segmax[seg]); denom[seg] += exp(...)
__global__ void k_attn_exp(float* __restrict__ score, const int* __restrict__ seg,
                           const float* __restrict__ segmax, float* __restrict__ denom, int cnt) {
  int tid = blockIdx.x * blockDim.x + threadIdx.x;
  if (tid >= cnt * 8) return;
  int e = tid >> 3, hh = tid & 7;
  int s = seg[e];
  float x = expf(score[tid] - segmax[(size_t)s * 8 + hh]);
  score[tid] = x;
  atomicAdd(&denom[(size_t)s * 8 + hh], x);
}

__global__ void k_attn_apply_node(const float* __restrict__ w, const float* __restrict__ denom,
                                  const float* __restrict__ eV, const float* __restrict__ nV,
                                  const int* __restrict__ u, const int* __restrict__ v,
                                  float* __restrict__ upd, int E) {
  int tid = blockIdx.x * blockDim.x + threadIdx.x;
  if (tid >= E * 8) return;
  int e = tid >> 3, hh = tid & 7;
  int s = u[e], d = v[e];
  float wn = w[tid] / denom[(size_t)s * 8 + hh];
  const float* v1 = eV + (size_t)e * 128 + hh * 16;
  const float* v2 = nV + (size_t)d * 128 + hh * 16;
  float* o = upd + (size_t)s * 128 + hh * 16;
#pragma unroll
  for (int i = 0; i < 16; ++i) atomicAdd(&o[i], wn * (v1[i] + v2[i]));
}

__global__ void k_attn_apply_pair(const float* __restrict__ w, const float* __restrict__ denom,
                                  const float* __restrict__ Ve,
                                  const int* __restrict__ ps, const int* __restrict__ pd,
                                  float* __restrict__ upd, int P) {
  int tid = blockIdx.x * blockDim.x + threadIdx.x;
  if (tid >= P * 8) return;
  int p = tid >> 3, hh = tid & 7;
  int a = ps[p], b = pd[p];
  float wn = w[tid] / denom[(size_t)a * 8 + hh];
  const float* vv = Ve + (size_t)b * 128 + hh * 16;
  float* o = upd + (size_t)a * 128 + hh * 16;
#pragma unroll
  for (int i = 0; i < 16; ++i) atomicAdd(&o[i], wn * vv[i]);
}

// ============================ host ============================

extern "C" void kernel_launch(void* const* d_in, const int* in_sizes, int n_in,
                              void* d_out, int out_size, void* d_ws, size_t ws_size,
                              hipStream_t stream) {
  (void)n_in; (void)out_size; (void)ws_size;
  int idx = 0;
  const float* in_n1 = (const float*)d_in[idx++];
  const float* in_n2 = (const float*)d_in[idx++];
  const float* in_e1 = (const float*)d_in[idx++];
  const float* in_e2 = (const float*)d_in[idx++];

  struct Lin { const float* b; const float* w; };
  struct LNp { const float* b; const float* g; };
  struct MHAp { Lin Wek, Weq, Wev, Wnk, Wnq, Wnv, Wo; };
  struct LayerP { MHAp emha; Lin efc1, efc2; LNp ln_e1, ln_e2, ln_effn, ln_n1, ln_n2, ln_nffn;
                  Lin nfc1, nfc2; MHAp nmha; };

  auto nextF   = [&]() { return (const float*)d_in[idx++]; };
  auto nextLin = [&]() { Lin l; l.b = nextF(); l.w = nextF(); return l; };          // keys: b, w
  auto nextLN  = [&]() { LNp l; l.b = nextF(); l.g = nextF(); return l; };          // keys: b, g
  auto nextMHA = [&]() { MHAp m; m.Wek = nextLin(); m.Weq = nextLin(); m.Wev = nextLin();
                         m.Wnk = nextLin(); m.Wnq = nextLin(); m.Wnv = nextLin();
                         m.Wo = nextLin(); return m; };
  auto nextLayer = [&]() { LayerP L;
    L.emha = nextMHA(); L.efc1 = nextLin(); L.efc2 = nextLin();
    L.ln_e1 = nextLN(); L.ln_e2 = nextLN(); L.ln_effn = nextLN();
    L.ln_n1 = nextLN(); L.ln_n2 = nextLN(); L.ln_nffn = nextLN();
    L.nfc1 = nextLin(); L.nfc2 = nextLin(); L.nmha = nextMHA(); return L; };

  // params pytree flattened in sorted-key order:
  Lin edge_dec  = nextLin();
  Lin edge_enc1 = nextLin();
  Lin edge_enc2 = nextLin();
  LayerP layers[2];
  layers[0] = nextLayer();
  layers[1] = nextLayer();
  Lin node_dec  = nextLin();
  Lin node_enc1 = nextLin();
  Lin node_enc2 = nextLin();

  const int* edge_index = (const int*)d_in[idx++];
  int nsrc_idx = idx;
  const int* nsrc = (const int*)d_in[idx++];
  const int* ndst = (const int*)d_in[idx++];

  const int Nn = in_sizes[0] / 2;       // 10000 nodes
  const int Ee = in_sizes[2] / 2;       // 20000 edges
  const int Pp = in_sizes[nsrc_idx];    // number of edge pairs
  const int* u = edge_index;            // edge_index[0]
  const int* v = edge_index + Ee;       // edge_index[1]

  // ---------------- workspace bump allocation ----------------
  char* base = (char*)d_ws;
  size_t off = 0;
  auto alloc  = [&](size_t bytes) -> void* { void* p = base + off; off = (off + bytes + 255) & ~(size_t)255; return p; };
  auto allocF = [&](size_t n) { return (float*)alloc(n * sizeof(float)); };

  __bf16* wpack = (__bf16*)alloc((size_t)3072 * 128 * sizeof(__bf16));
  float* nt  = allocF((size_t)Nn * 128);
  float* et  = allocF((size_t)Ee * 128);
  float* row = allocF((size_t)Ee * 128);
  float* nQb = allocF((size_t)Nn * 128);
  float* nKb = allocF((size_t)Nn * 128);
  float* nVb = allocF((size_t)Nn * 128);
  float* eQb = allocF((size_t)Ee * 128);
  float* eKb = allocF((size_t)Ee * 128);
  float* eVb = allocF((size_t)Ee * 128);
  float* Qe  = allocF((size_t)Ee * 128);
  float* Ke  = allocF((size_t)Ee * 128);
  float* Ve  = allocF((size_t)Ee * 128);
  float* sbuf   = allocF((size_t)(Pp > Ee ? Pp : Ee) * 8);
  float* segmax = allocF((size_t)Ee * 8);
  float* denom  = allocF((size_t)Ee * 8);
  float* upd = allocF((size_t)Ee * 128);
  float* npb = allocF((size_t)Nn * 128);
  float* epb = allocF((size_t)Ee * 128);
  float* hbf = allocF((size_t)Ee * 512);

  // ---------------- launch helpers ----------------
  auto doGemm = [&](const float* Aact, const float* W, const float* bias, int M, int K, int Ncols,
                    const float* addend, float* out, int relu) {
    int tw = (Ncols / 16) * (K / 32) * 32;
    k_pack_b<<<div_up(tw, TPB), TPB, 0, stream>>>(W, wpack, K, Ncols);
    int waves  = (M / 16) * (Ncols / 128);
    int blocks = div_up(waves * 32, TPB);
    if (K == 3072)
      k_gemm_wmma<96><<<blocks, TPB, 0, stream>>>(Aact, wpack, bias, addend, out, M, Ncols, relu);
    else if (K == 512)
      k_gemm_wmma<16><<<blocks, TPB, 0, stream>>>(Aact, wpack, bias, addend, out, M, Ncols, relu);
    else
      k_gemm_wmma<4><<<blocks, TPB, 0, stream>>>(Aact, wpack, bias, addend, out, M, Ncols, relu);
  };
  auto doLN = [&](const float* X, LNp p, float* Y, int M) {
    k_layernorm<<<div_up(M * 32, TPB), TPB, 0, stream>>>(X, p.g, p.b, Y, M);
  };

  // ---------------- encoders ----------------
  k_lin2<<<div_up(Nn * 128, TPB), TPB, 0, stream>>>(in_n1, node_enc1.w, node_enc1.b, nt, Nn, 128);
  doGemm(in_n2, node_enc2.w, node_enc2.b, Nn, 3072, 128, nt, nt, 0);

  k_lin2<<<div_up(Ee * 128, TPB), TPB, 0, stream>>>(in_e1, edge_enc1.w, edge_enc1.b, et, Ee, 128);
  doGemm(in_e2, edge_enc2.w, edge_enc2.b, Ee, 3072, 128, et, et, 0);

  // ---------------- transformer layers ----------------
  for (int li = 0; li < 2; ++li) {
    LayerP& L = layers[li];

    // ---- node MHA ----
    doLN(nt, L.ln_n1, row, Nn);
    doGemm(row, L.nmha.Wnq.w, L.nmha.Wnq.b, Nn, 128, 128, nullptr, nQb, 0);
    doGemm(row, L.nmha.Wnk.w, L.nmha.Wnk.b, Nn, 128, 128, nullptr, nKb, 0);
    doGemm(row, L.nmha.Wnv.w, L.nmha.Wnv.b, Nn, 128, 128, nullptr, nVb, 0);
    doLN(et, L.ln_e1, row, Ee);
    doGemm(row, L.nmha.Weq.w, L.nmha.Weq.b, Ee, 128, 128, nullptr, eQb, 0);
    doGemm(row, L.nmha.Wek.w, L.nmha.Wek.b, Ee, 128, 128, nullptr, eKb, 0);
    doGemm(row, L.nmha.Wev.w, L.nmha.Wev.b, Ee, 128, 128, nullptr, eVb, 0);

    k_fill<<<div_up(Nn * 8, TPB), TPB, 0, stream>>>(segmax, -3.0e38f, Nn * 8);
    hipMemsetAsync(denom, 0, (size_t)Nn * 8 * sizeof(float), stream);
    k_attn_score_node<<<div_up(Ee * 8, TPB), TPB, 0, stream>>>(eQb, eKb, nQb, nKb, u, v, sbuf, segmax, Ee);
    k_attn_exp<<<div_up(Ee * 8, TPB), TPB, 0, stream>>>(sbuf, u, segmax, denom, Ee);
    hipMemsetAsync(upd, 0, (size_t)Nn * 128 * sizeof(float), stream);
    k_attn_apply_node<<<div_up(Ee * 8, TPB), TPB, 0, stream>>>(sbuf, denom, eVb, nVb, u, v, upd, Ee);

    doGemm(upd, L.nmha.Wo.w, L.nmha.Wo.b, Nn, 128, 128, nt, npb, 0);   // np_ = attn + nt

    // ---- node FFN ----
    doLN(npb, L.ln_nffn, row, Nn);
    doGemm(row, L.nfc1.w, L.nfc1.b, Nn, 128, 512, nullptr, hbf, 1);    // relu
    doGemm(hbf, L.nfc2.w, L.nfc2.b, Nn, 512, 128, npb, nt, 0);         // nt = ffn + np_

    // ---- edge MHA ----
    doLN(nt, L.ln_n2, row, Nn);
    doGemm(row, L.emha.Wnq.w, L.emha.Wnq.b, Nn, 128, 128, nullptr, nQb, 0);
    doGemm(row, L.emha.Wnk.w, L.emha.Wnk.b, Nn, 128, 128, nullptr, nKb, 0);
    doGemm(row, L.emha.Wnv.w, L.emha.Wnv.b, Nn, 128, 128, nullptr, nVb, 0);
    doLN(et, L.ln_e2, row, Ee);
    doGemm(row, L.emha.Weq.w, L.emha.Weq.b, Ee, 128, 128, nullptr, eQb, 0);
    doGemm(row, L.emha.Wek.w, L.emha.Wek.b, Ee, 128, 128, nullptr, eKb, 0);
    doGemm(row, L.emha.Wev.w, L.emha.Wev.b, Ee, 128, 128, nullptr, eVb, 0);

    k_combine3<<<div_up(Ee * 128, TPB), TPB, 0, stream>>>(eQb, nQb, u, v, Qe, Ee);
    k_combine3<<<div_up(Ee * 128, TPB), TPB, 0, stream>>>(eKb, nKb, u, v, Ke, Ee);
    k_combine3<<<div_up(Ee * 128, TPB), TPB, 0, stream>>>(eVb, nVb, u, v, Ve, Ee);

    k_fill<<<div_up(Ee * 8, TPB), TPB, 0, stream>>>(segmax, -3.0e38f, Ee * 8);
    hipMemsetAsync(denom, 0, (size_t)Ee * 8 * sizeof(float), stream);
    k_attn_score_pair<<<div_up(Pp * 8, TPB), TPB, 0, stream>>>(Qe, Ke, nsrc, ndst, sbuf, segmax, Pp);
    k_attn_exp<<<div_up(Pp * 8, TPB), TPB, 0, stream>>>(sbuf, nsrc, segmax, denom, Pp);
    hipMemsetAsync(upd, 0, (size_t)Ee * 128 * sizeof(float), stream);
    k_attn_apply_pair<<<div_up(Pp * 8, TPB), TPB, 0, stream>>>(sbuf, denom, Ve, nsrc, ndst, upd, Pp);

    doGemm(upd, L.emha.Wo.w, L.emha.Wo.b, Ee, 128, 128, et, epb, 0);   // ep_ = attn + et

    // ---- edge FFN ----
    doLN(epb, L.ln_effn, row, Ee);
    doGemm(row, L.efc1.w, L.efc1.b, Ee, 128, 512, nullptr, hbf, 1);    // relu
    doGemm(hbf, L.efc2.w, L.efc2.b, Ee, 512, 128, epb, et, 0);         // et = ffn + ep_
  }

  // ---------------- decoders ----------------
  float* outp = (float*)d_out;
  doGemm(nt, node_dec.w, node_dec.b, Nn, 128, 128, nullptr, outp, 0);
  doGemm(et, edge_dec.w, edge_dec.b, Ee, 128, 128, nullptr, outp + (size_t)Nn * 128, 0);
}
